// _CausalSelfAttention_12300786336463
// MI455X (gfx1250) — matrix operand order repaired and checked by reading, not compiled, not run on hardware
//
#include <hip/hip_runtime.h>
#include <hip/hip_bf16.h>
#include <cstdint>

typedef __bf16 v16bf __attribute__((ext_vector_type(16)));
typedef __bf16 v8bf  __attribute__((ext_vector_type(8)));
typedef float  v8f   __attribute__((ext_vector_type(8)));
typedef int    v4i   __attribute__((vector_size(16)));   // matches builtin proto

#define LOG2E_F 1.4426950408889634f
#define NEG_BIG (-1e30f)

#if __has_builtin(__builtin_amdgcn_global_load_async_to_lds_b128)
#define HAVE_ASYNC_LDS 1
#endif

static __device__ __forceinline__ void async_copy_b128(void* lds_dst,
                                                       const void* gsrc) {
#ifdef HAVE_ASYNC_LDS
  __builtin_amdgcn_global_load_async_to_lds_b128(
      (__attribute__((address_space(1))) v4i*)gsrc,
      (__attribute__((address_space(3))) v4i*)lds_dst, 0, 0);
#else
  *(uint4*)lds_dst = *(const uint4*)gsrc;
#endif
}

template <int N>
static __device__ __forceinline__ void wait_async() {
#ifdef HAVE_ASYNC_LDS
#if __has_builtin(__builtin_amdgcn_s_wait_asynccnt)
  __builtin_amdgcn_s_wait_asynccnt(N);
#else
  if constexpr (N == 0)      asm volatile("s_wait_asynccnt 0x0" ::: "memory");
  else if constexpr (N == 2) asm volatile("s_wait_asynccnt 0x2" ::: "memory");
  else if constexpr (N == 4) asm volatile("s_wait_asynccnt 0x4" ::: "memory");
  else                       asm volatile("s_wait_asynccnt 0x0" ::: "memory");
#endif
#endif
}

static __device__ __forceinline__ v8f v8f_zero() {
  v8f z = {0.f, 0.f, 0.f, 0.f, 0.f, 0.f, 0.f, 0.f};
  return z;
}
static __device__ __forceinline__ v16bf cat16(v8bf lo, v8bf hi) {
  return __builtin_shufflevector(lo, hi, 0,1,2,3,4,5,6,7,8,9,10,11,12,13,14,15);
}
static __device__ __forceinline__ v8f wmma_bf16(v16bf a, v16bf b, v8f c) {
  return __builtin_amdgcn_wmma_f32_16x16x32_bf16(false, a, false, b, (short)0, c,
                                                 false, false);
}

// ---------------------------------------------------------------- fp32->bf16
__global__ void __launch_bounds__(256)
cvt_f32_bf16(const float* __restrict__ in, __bf16* __restrict__ out, int n4) {
  int i = blockIdx.x * 256 + threadIdx.x;
  if (i < n4) {
    float4 v = ((const float4*)in)[i];
    union { __bf16 b[4]; uint2 u; } o;
    o.b[0] = (__bf16)v.x; o.b[1] = (__bf16)v.y;
    o.b[2] = (__bf16)v.z; o.b[3] = (__bf16)v.w;
    ((uint2*)out)[i] = o.u;
  }
}

// ------------------------------------------------------- bf16 WMMA GEMM
// C = A[MxK] * B[KxN] + bias.  MODE 0: fp32 row-major out.
// MODE 1: scatter bf16 into Q/K/V [B*H][T][64] head layout (N == 2304).
// 2-stage pipeline: A tile via async global->LDS; B tile via register staging.
template <int MODE>
__global__ void __launch_bounds__(256)
gemm_bf16_wmma(const __bf16* __restrict__ A, const __bf16* __restrict__ Bm,
               const float* __restrict__ bias, float* __restrict__ outF,
               __bf16* __restrict__ Qo, __bf16* __restrict__ Ko,
               __bf16* __restrict__ Vo, int M, int N, int K) {
  __shared__ __align__(32) __bf16 As[2][128 * 32];   // [m][k]
  __shared__ __align__(32) __bf16 Bts[2][64 * 32];   // [n][k] (transposed)

  const int tid = threadIdx.x;
  const int wv = tid >> 5, lane = tid & 31;
  const int hf = lane >> 4, l16 = lane & 15;
  const int wm = wv >> 1, wn = wv & 1;               // 4x2 wave grid
  const int bm = blockIdx.y * 128;
  const int bn = blockIdx.x * 64;
  const int nsteps = K >> 5;

  // A-tile async issue: 2 x b128 per thread per stage
  auto issueA = [&](int step, int buf) {
    #pragma unroll
    for (int i = 0; i < 2; ++i) {
      int id = tid + i * 256;
      int row = id >> 2;
      int c8 = (id & 3) << 3;
      async_copy_b128(&As[buf][row * 32 + c8],
                      &A[(size_t)(bm + row) * K + step * 32 + c8]);
    }
  };
  const int kr = tid >> 3;            // 0..31
  const int nb = (tid & 7) << 3;      // 0..56
  auto loadB = [&](int step) -> uint4 {
    return *(const uint4*)&Bm[(size_t)(step * 32 + kr) * N + bn + nb];
  };

  v8f acc[2][2];
  #pragma unroll
  for (int i = 0; i < 2; ++i)
    #pragma unroll
    for (int j = 0; j < 2; ++j) acc[i][j] = v8f_zero();

  issueA(0, 0);
  uint4 breg = loadB(0);

  for (int step = 0; step < nsteps; ++step) {
    const int cur = step & 1;
    const bool has_next = (step + 1 < nsteps);
    uint4 bnext = {};
    if (has_next) {
      issueA(step + 1, cur ^ 1);
      bnext = loadB(step + 1);
      if (step + 2 < nsteps) {        // pull stage+2 lines toward L2
        __builtin_prefetch(&A[(size_t)(bm + (tid >> 1)) * K + (step + 2) * 32],
                           0, 0);
        __builtin_prefetch(&Bm[(size_t)((step + 2) * 32 + kr) * N + bn + nb],
                           0, 0);
      }
    }
    {                                 // transposed store of current B stage
      const __bf16* tp = (const __bf16*)&breg;
      #pragma unroll
      for (int e = 0; e < 8; ++e) Bts[cur][(nb + e) * 32 + kr] = tp[e];
    }
    if (has_next) wait_async<2>(); else wait_async<0>();
    __syncthreads();

    v16bf af[2], bfv[2];
    #pragma unroll
    for (int i = 0; i < 2; ++i) {
      int row = wm * 32 + i * 16 + l16;
      af[i] = *(const v16bf*)&As[cur][row * 32 + hf * 16];
    }
    #pragma unroll
    for (int j = 0; j < 2; ++j) {
      int col = wn * 32 + j * 16 + l16;
      bfv[j] = *(const v16bf*)&Bts[cur][col * 32 + hf * 16];
    }
    #pragma unroll
    for (int i = 0; i < 2; ++i)
      #pragma unroll
      for (int j = 0; j < 2; ++j)
        acc[i][j] = wmma_bf16(af[i], bfv[j], acc[i][j]);

    __syncthreads();
    if (has_next) breg = bnext;
  }

  #pragma unroll
  for (int i = 0; i < 2; ++i)
    #pragma unroll
    for (int j = 0; j < 2; ++j)
      #pragma unroll
      for (int r = 0; r < 8; ++r) {
        int m = bm + wm * 32 + i * 16 + hf * 8 + r;
        int n = bn + wn * 32 + j * 16 + l16;
        float v = acc[i][j][r] + bias[n];
        if (MODE == 0) {
          outF[(size_t)m * N + n] = v;
        } else {
          int which = n / 768, hh = (n % 768) >> 6, d = n & 63;
          int bb = m >> 11, t = m & 2047;
          size_t o = (((size_t)bb * 12 + hh) * 2048 + t) * 64 + d;
          __bf16* dst = (which == 0) ? Qo : ((which == 1) ? Ko : Vo);
          dst[o] = (__bf16)v;
        }
      }
}

// ------------------------------------------------------- flash attention
// grid: (32 q-blocks, 24 b*h). 4 waves, 16 q-rows per wave, Dh=64, kv-block=64.
// 2-stage pipeline: K tile async global->LDS, V tile staged through registers.
__global__ void __launch_bounds__(128)
flash_attn_kernel(const __bf16* __restrict__ Qg, const __bf16* __restrict__ Kg,
                  const __bf16* __restrict__ Vg, __bf16* __restrict__ Yb) {
  __shared__ __align__(32) __bf16 Qs[64 * 64];        // [qrow][d]
  __shared__ __align__(32) __bf16 Ks[2][64 * 64];     // [krow][d]
  __shared__ __align__(32) __bf16 Vts[2][64 * 64];    // [d][krow]
  __shared__ __align__(32) __bf16 Ps[4][16 * 64];     // per-wave P [m][k]

  const int tid = threadIdx.x;
  const int wv = tid >> 5, lane = tid & 31;
  const int hf = lane >> 4, l16 = lane & 15;
  const int bh = blockIdx.y;
  const int qb = blockIdx.x;

  const __bf16* Qbase = Qg + ((size_t)bh * 2048 + (size_t)qb * 64) * 64;
  #pragma unroll
  for (int i = 0; i < 4; ++i) {
    int id = tid + i * 128;
    int row = id >> 3;
    int c8 = (id & 7) << 3;
    *(uint4*)&Qs[row * 64 + c8] = *(const uint4*)&Qbase[(size_t)row * 64 + c8];
  }

  auto issueK = [&](int j, int buf) {     // 4 x b128 per thread per stage
    #pragma unroll
    for (int i = 0; i < 4; ++i) {
      int id = tid + i * 128;
      int row = id >> 3;
      int c8 = (id & 7) << 3;
      async_copy_b128(&Ks[buf][row * 64 + c8],
                      &Kg[((size_t)bh * 2048 + (size_t)j * 64 + row) * 64 + c8]);
    }
  };
  auto loadV = [&](int j, uint4 vr[4]) {
    #pragma unroll
    for (int i = 0; i < 4; ++i) {
      int id = tid + i * 128;
      int row = id >> 3;
      int c8 = (id & 7) << 3;
      vr[i] = *(const uint4*)&Vg[((size_t)bh * 2048 + (size_t)j * 64 + row) * 64 + c8];
    }
  };

  float m_i[8], l_i[8];
  v8f accO[4];
  #pragma unroll
  for (int r = 0; r < 8; ++r) { m_i[r] = NEG_BIG; l_i[r] = 0.f; }
  #pragma unroll
  for (int d = 0; d < 4; ++d) accO[d] = v8f_zero();

  uint4 vr[4], vrn[4];
  issueK(0, 0);
  loadV(0, vr);

  for (int j = 0; j <= qb; ++j) {
    const int cur = j & 1;
    const bool has_next = (j < qb);
    if (has_next) {
      issueK(j + 1, cur ^ 1);
      loadV(j + 1, vrn);
      if (j + 2 <= qb) {
        size_t p2 = ((size_t)bh * 2048 + (size_t)(j + 2) * 64 + (tid >> 1)) * 64;
        __builtin_prefetch(&Kg[p2], 0, 0);
        __builtin_prefetch(&Vg[p2], 0, 0);
      }
    }
    {                                   // transposed scatter of current V stage
      #pragma unroll
      for (int i = 0; i < 4; ++i) {
        int id = tid + i * 128;
        int row = id >> 3;
        int c8 = (id & 7) << 3;
        const __bf16* tp = (const __bf16*)&vr[i];
        #pragma unroll
        for (int e = 0; e < 8; ++e) Vts[cur][(c8 + e) * 64 + row] = tp[e];
      }
    }
    if (has_next) wait_async<4>(); else wait_async<0>();
    __syncthreads();

    // S = Q * K^T  (16 rows x 64 cols per wave)
    v8f s[4];
    #pragma unroll
    for (int c = 0; c < 4; ++c) s[c] = v8f_zero();
    #pragma unroll
    for (int kk = 0; kk < 2; ++kk) {
      int qr = wv * 16 + l16;
      v16bf af = *(const v16bf*)&Qs[qr * 64 + kk * 32 + hf * 16];
      #pragma unroll
      for (int c = 0; c < 4; ++c) {
        v16bf bfr = *(const v16bf*)&Ks[cur][(c * 16 + l16) * 64 + kk * 32 + hf * 16];
        s[c] = wmma_bf16(af, bfr, s[c]);
      }
    }

    const float sc = 0.125f;  // 1/sqrt(64)
    if (j == qb) {
      #pragma unroll
      for (int c = 0; c < 4; ++c)
        #pragma unroll
        for (int r = 0; r < 8; ++r) {
          int rr = wv * 16 + hf * 8 + r;
          int cc = c * 16 + l16;
          s[c][r] = (cc > rr) ? NEG_BIG : s[c][r] * sc;
        }
    } else {
      #pragma unroll
      for (int c = 0; c < 4; ++c)
        #pragma unroll
        for (int r = 0; r < 8; ++r) s[c][r] *= sc;
    }

    // streaming softmax (rows live in half-waves; width-16 shuffles)
    #pragma unroll
    for (int r = 0; r < 8; ++r) {
      float mx = fmaxf(fmaxf(s[0][r], s[1][r]), fmaxf(s[2][r], s[3][r]));
      #pragma unroll
      for (int o = 1; o < 16; o <<= 1) mx = fmaxf(mx, __shfl_xor(mx, o, 16));
      float mnew = fmaxf(m_i[r], mx);
      float alpha = exp2f((m_i[r] - mnew) * LOG2E_F);
      float p[4], ps = 0.f;
      #pragma unroll
      for (int c = 0; c < 4; ++c) {
        p[c] = exp2f((s[c][r] - mnew) * LOG2E_F);
        ps += p[c];
      }
      #pragma unroll
      for (int o = 1; o < 16; o <<= 1) ps += __shfl_xor(ps, o, 16);
      l_i[r] = l_i[r] * alpha + ps;
      m_i[r] = mnew;
      #pragma unroll
      for (int d = 0; d < 4; ++d) accO[d][r] *= alpha;
      #pragma unroll
      for (int c = 0; c < 4; ++c)
        Ps[wv][(hf * 8 + r) * 64 + c * 16 + l16] = (__bf16)p[c];
    }

    // O += P * V   (same-wave LDS C-layout -> A-layout round trip; DS in-order)
    #pragma unroll
    for (int kk = 0; kk < 2; ++kk) {
      v16bf af = *(const v16bf*)&Ps[wv][l16 * 64 + kk * 32 + hf * 16];
      #pragma unroll
      for (int dt = 0; dt < 4; ++dt) {
        v16bf bfr = *(const v16bf*)&Vts[cur][(dt * 16 + l16) * 64 + kk * 32 + hf * 16];
        accO[dt] = wmma_bf16(af, bfr, accO[dt]);
      }
    }
    __syncthreads();
    if (has_next) {
      #pragma unroll
      for (int i = 0; i < 4; ++i) vr[i] = vrn[i];
    }
  }

  // epilogue: normalize and write y in [B*T][C] layout for the proj GEMM
  const int b = bh / 12, h = bh % 12;
  #pragma unroll
  for (int dt = 0; dt < 4; ++dt)
    #pragma unroll
    for (int r = 0; r < 8; ++r) {
      float v = accO[dt][r] / l_i[r];
      int t = qb * 64 + wv * 16 + hf * 8 + r;
      int col = h * 64 + dt * 16 + l16;
      Yb[((size_t)b * 2048 + t) * 768 + col] = (__bf16)v;
    }
}

// ---------------------------------------------------------------- launch
extern "C" void kernel_launch(void* const* d_in, const int* in_sizes, int n_in,
                              void* d_out, int out_size, void* d_ws,
                              size_t ws_size, hipStream_t stream) {
  (void)in_sizes; (void)n_in; (void)out_size; (void)ws_size;
  const float* x      = (const float*)d_in[0];
  const float* w_qkv  = (const float*)d_in[1];
  const float* b_qkv  = (const float*)d_in[2];
  const float* w_proj = (const float*)d_in[3];
  const float* b_proj = (const float*)d_in[4];
  float* out = (float*)d_out;

  const int M = 2 * 2048;   // B*T = 4096
  const int C = 768;
  const int N1 = 3 * C;     // 2304
  const int BH = 24;

  char* ws = (char*)d_ws;
  size_t off = 0;
  auto alloc = [&](size_t bytes) -> char* {
    char* p = ws + off;
    off += (bytes + 255) & ~(size_t)255;
    return p;
  };
  __bf16* xb    = (__bf16*)alloc((size_t)M * C * 2);
  __bf16* wqkvb = (__bf16*)alloc((size_t)C * N1 * 2);
  __bf16* wprjb = (__bf16*)alloc((size_t)C * C * 2);
  __bf16* Qb    = (__bf16*)alloc((size_t)BH * 2048 * 64 * 2);
  __bf16* Kb    = (__bf16*)alloc((size_t)BH * 2048 * 64 * 2);
  __bf16* Vb    = (__bf16*)alloc((size_t)BH * 2048 * 64 * 2);
  __bf16* yb    = (__bf16*)alloc((size_t)M * C * 2);

  // 1) fp32 -> bf16 conversions
  {
    int n4 = M * C / 4;
    cvt_f32_bf16<<<(n4 + 255) / 256, 256, 0, stream>>>(x, xb, n4);
    n4 = C * N1 / 4;
    cvt_f32_bf16<<<(n4 + 255) / 256, 256, 0, stream>>>(w_qkv, wqkvb, n4);
    n4 = C * C / 4;
    cvt_f32_bf16<<<(n4 + 255) / 256, 256, 0, stream>>>(w_proj, wprjb, n4);
  }

  // 2) QKV GEMM -> per-head Q/K/V bf16
  {
    dim3 grid(N1 / 64, M / 128);
    gemm_bf16_wmma<1><<<grid, 256, 0, stream>>>(xb, wqkvb, b_qkv, nullptr, Qb,
                                                Kb, Vb, M, N1, C);
  }

  // 3) flash attention -> yb [M][C] bf16
  {
    dim3 grid(2048 / 64, BH);
    flash_attn_kernel<<<grid, 128, 0, stream>>>(Qb, Kb, Vb, yb);
  }

  // 4) output projection -> fp32 d_out
  {
    dim3 grid(C / 64, M / 128);
    gemm_bf16_wmma<0><<<grid, 256, 0, stream>>>(yb, wprjb, b_proj, out, nullptr,
                                                nullptr, nullptr, M, C, C);
  }
}
